// CrossAttentionModulation_25409026523945
// MI455X (gfx1250) — matrix operand order repaired and checked by reading, not compiled, not run on hardware
//
#include <hip/hip_runtime.h>
#include <hip/hip_bf16.h>

typedef __attribute__((ext_vector_type(16))) _Float16 v16h;
typedef __attribute__((ext_vector_type(8)))  _Float16 v8h;
typedef __attribute__((ext_vector_type(8)))  float    v8f;

#define WMMA16(a, b, c) __builtin_amdgcn_wmma_f32_16x16x32_f16( \
    false, (a), false, (b), (short)0, (c), false, false)

constexpr int C_DIM = 1024;
constexpr int NHEAD = 16;
constexpr int DHEAD = 64;
constexpr int LSEQ  = 4096;
constexpr int LP    = 256;
constexpr int BATCH = 8;

__device__ inline v16h cat8(v8h a, v8h b) {
  return __builtin_shufflevector(a, b, 0,1,2,3,4,5,6,7,8,9,10,11,12,13,14,15);
}

// Build a WMMA A-operand chunk (32 K-values) from f32 row with layernorm applied.
// k order used here (contiguous, same as B): halves 0..7 = K[off .. off+7], halves 8..15 = K[off+8 .. off+15]
__device__ inline v16h norm16(const float* __restrict__ xr, int off, float mu, float rs) {
  const float4* p = (const float4*)(xr + off);
  float4 f0 = p[0], f1 = p[1];
  const float4* q = (const float4*)(xr + off + 8);
  float4 f2 = q[0], f3 = q[1];
  v16h a;
  a[0]  = (_Float16)((f0.x - mu) * rs); a[1]  = (_Float16)((f0.y - mu) * rs);
  a[2]  = (_Float16)((f0.z - mu) * rs); a[3]  = (_Float16)((f0.w - mu) * rs);
  a[4]  = (_Float16)((f1.x - mu) * rs); a[5]  = (_Float16)((f1.y - mu) * rs);
  a[6]  = (_Float16)((f1.z - mu) * rs); a[7]  = (_Float16)((f1.w - mu) * rs);
  a[8]  = (_Float16)((f2.x - mu) * rs); a[9]  = (_Float16)((f2.y - mu) * rs);
  a[10] = (_Float16)((f2.z - mu) * rs); a[11] = (_Float16)((f2.w - mu) * rs);
  a[12] = (_Float16)((f3.x - mu) * rs); a[13] = (_Float16)((f3.y - mu) * rs);
  a[14] = (_Float16)((f3.z - mu) * rs); a[15] = (_Float16)((f3.w - mu) * rs);
  return a;
}

// ---------------- per-row mean / rsigma ----------------
__global__ void rowstats_kernel(const float* __restrict__ X, float* __restrict__ stats, int rows) {
  int wave = threadIdx.x >> 5, lane = threadIdx.x & 31;
  int row = blockIdx.x * 8 + wave;
  if (row >= rows) return;
  const float4* xp = (const float4*)(X + (size_t)row * C_DIM);
  float s = 0.f, s2 = 0.f;
  #pragma unroll
  for (int j = 0; j < 8; ++j) {
    float4 v = xp[lane + 32 * j];
    s  += v.x + v.y + v.z + v.w;
    s2 += v.x * v.x + v.y * v.y + v.z * v.z + v.w * v.w;
  }
  #pragma unroll
  for (int m = 16; m >= 1; m >>= 1) { s += __shfl_xor(s, m); s2 += __shfl_xor(s2, m); }
  if (lane == 0) {
    float mu  = s * (1.f / 1024.f);
    float var = s2 * (1.f / 1024.f) - mu * mu;
    stats[row * 2]     = mu;
    stats[row * 2 + 1] = rsqrtf(var + 1e-5f);
  }
}

// ---------------- fold LN weights into W, produce f16 W and fused bias ----------------
__global__ void packw_kernel(const float* __restrict__ W, const float* __restrict__ lnw,
                             const float* __restrict__ lnb, const float* __restrict__ bias,
                             _Float16* __restrict__ W16, float* __restrict__ biasP) {
  int wave = threadIdx.x >> 5, lane = threadIdx.x & 31;
  int n = blockIdx.x * 8 + wave;
  const float* wr = W + (size_t)n * C_DIM;
  _Float16* wo = W16 + (size_t)n * C_DIM;
  float acc = 0.f;
  for (int j = 0; j < 32; ++j) {
    int k = lane + 32 * j;
    float w = wr[k];
    acc += lnb[k] * w;
    wo[k] = (_Float16)(w * lnw[k]);
  }
  #pragma unroll
  for (int m = 16; m >= 1; m >>= 1) acc += __shfl_xor(acc, m);
  if (lane == 0) biasP[n] = bias[n] + acc;
}

// ---------------- V transpose: x[b,l,h*64+d] -> VT[(b*16+h)*64+d][l] (f16) ----------------
__global__ void packv_kernel(const float* __restrict__ X, _Float16* __restrict__ VT) {
  __shared__ __attribute__((aligned(16))) _Float16 tile[64][68];
  int bid = blockIdx.x;                  // b*1024 + h*64 + lt
  int lt = bid & 63, h = (bid >> 6) & 15, b = bid >> 10;
  int l0 = lt * 64;
  const float* xp = X + (size_t)b * LSEQ * C_DIM + h * DHEAD;
  #pragma unroll 4
  for (int rep = 0; rep < 16; ++rep) {
    int e = threadIdx.x + 256 * rep;
    int li = e >> 6, d = e & 63;
    tile[d][li] = (_Float16)xp[(size_t)(l0 + li) * C_DIM + d];
  }
  __syncthreads();
  _Float16* vp = VT + (size_t)(b * NHEAD + h) * DHEAD * LSEQ;
  #pragma unroll 4
  for (int rep = 0; rep < 16; ++rep) {
    int e = threadIdx.x + 256 * rep;
    int d = e >> 6, li = e & 63;
    vp[(size_t)d * LSEQ + l0 + li] = tile[d][li];
  }
}

// ---------------- fused LN + GEMM (Out = clip(LN(X) @ W16^T + biasP)) -> f16 ----------------
__global__ __launch_bounds__(256) void lngemm_kernel(
    const float* __restrict__ X, const float* __restrict__ stats,
    const _Float16* __restrict__ W16, const float* __restrict__ biasP,
    _Float16* __restrict__ Out) {
  int bn = blockIdx.x & 7, bm = blockIdx.x >> 3;
  int wave = threadIdx.x >> 5, lane = threadIdx.x & 31;
  int lo = lane & 15, hi = lane >> 4;
  int mw = wave & 3, nw = wave >> 2;
  int M0 = bm * 128 + mw * 32;     // 2 m-tiles of 16
  int N0 = bn * 128 + nw * 64;     // 4 n-tiles of 16

  int r0 = M0 + lo, r1 = M0 + 16 + lo;
  float mu0 = stats[r0 * 2], rs0 = stats[r0 * 2 + 1];
  float mu1 = stats[r1 * 2], rs1 = stats[r1 * 2 + 1];
  const float* x0 = X + (size_t)r0 * C_DIM;
  const float* x1 = X + (size_t)r1 * C_DIM;

  v8f acc[2][4] = {};
  for (int kc = 0; kc < 32; ++kc) {
    int K0 = kc * 32;
    v16h a0 = norm16(x0, K0 + hi * 16, mu0, rs0);
    v16h a1 = norm16(x1, K0 + hi * 16, mu1, rs1);
    #pragma unroll
    for (int nt = 0; nt < 4; ++nt) {
      int n = N0 + nt * 16 + lo;
      v16h bv = *(const v16h*)(W16 + (size_t)n * C_DIM + K0 + hi * 16);
      acc[0][nt] = WMMA16(a0, bv, acc[0][nt]);
      acc[1][nt] = WMMA16(a1, bv, acc[1][nt]);
    }
  }
  #pragma unroll
  for (int nt = 0; nt < 4; ++nt) {
    int n = N0 + nt * 16 + lo;
    float bs = biasP[n];
    #pragma unroll
    for (int mt = 0; mt < 2; ++mt) {
      int rbase = M0 + mt * 16 + 8 * hi;
      #pragma unroll
      for (int i = 0; i < 8; ++i) {
        float v = acc[mt][nt][i] + bs;
        v = fminf(fmaxf(v, -10.f), 10.f);
        Out[(size_t)(rbase + i) * C_DIM + n] = (_Float16)v;
      }
    }
  }
}

// ---------------- flash attention: per (b,h,query-half), accumulate mean_q into gsum ----------------
__global__ __launch_bounds__(256) void attn_kernel(
    const _Float16* __restrict__ Q16, const _Float16* __restrict__ K16,
    const _Float16* __restrict__ VT, float* __restrict__ gsum) {
  __shared__ __attribute__((aligned(16))) _Float16 pbuf[8][16][72];
  int bh = blockIdx.x >> 1, qh = blockIdx.x & 1;
  int b = bh >> 4, h = bh & 15;
  int wave = threadIdx.x >> 5, lane = threadIdx.x & 31;
  int lo = lane & 15, hi = lane >> 4;

  int q = qh * 128 + wave * 16 + lo;
  const _Float16* qp = Q16 + (size_t)(b * LP + q) * C_DIM + h * DHEAD;
  v16h aq0 = cat8(*(const v8h*)(qp + hi * 16),      *(const v8h*)(qp + 8 + hi * 16));
  v16h aq1 = cat8(*(const v8h*)(qp + 32 + hi * 16), *(const v8h*)(qp + 40 + hi * 16));

  float mrow[8], lrow[8];
  v8f o[4] = {};
  #pragma unroll
  for (int i = 0; i < 8; ++i) { mrow[i] = -3.0e38f; lrow[i] = 0.f; }
  const float cexp = 0.125f * 1.44269504088896f;  // scale * log2(e)

  const _Float16* kbase = K16 + (size_t)b * LSEQ * C_DIM + h * DHEAD;
  const _Float16* vbase = VT + (size_t)bh * DHEAD * LSEQ;

  for (int kt = 0; kt < 64; ++kt) {
    int kb = kt * 64;
    if (kt < 63)
      __builtin_prefetch(kbase + (size_t)(kb + 64 + lane * 2) * C_DIM, 0, 3);
    // scores S = Q K^T
    v8f s[4];
    #pragma unroll
    for (int nt = 0; nt < 4; ++nt) {
      const _Float16* kp = kbase + (size_t)(kb + nt * 16 + lo) * C_DIM;
      v16h bk0 = *(const v16h*)(kp + hi * 16);
      v16h bk1 = *(const v16h*)(kp + 32 + hi * 16);
      v8f z = {};
      z = WMMA16(aq0, bk0, z);
      s[nt] = WMMA16(aq1, bk1, z);
    }
    // online softmax over the 64-key tile (row stats across 16-lane halves)
    float sf[8];
    #pragma unroll
    for (int i = 0; i < 8; ++i) {
      float t = fmaxf(fmaxf(s[0][i], s[1][i]), fmaxf(s[2][i], s[3][i]));
      t = fmaxf(t, __shfl_xor(t, 1)); t = fmaxf(t, __shfl_xor(t, 2));
      t = fmaxf(t, __shfl_xor(t, 4)); t = fmaxf(t, __shfl_xor(t, 8));
      float mn = fmaxf(mrow[i], t);
      sf[i] = exp2f(cexp * (mrow[i] - mn));
      mrow[i] = mn;
      float rs = 0.f;
      #pragma unroll
      for (int nt = 0; nt < 4; ++nt) {
        float p = exp2f(cexp * (s[nt][i] - mn));
        s[nt][i] = p;
        rs += p;
      }
      rs += __shfl_xor(rs, 1); rs += __shfl_xor(rs, 2);
      rs += __shfl_xor(rs, 4); rs += __shfl_xor(rs, 8);
      lrow[i] = lrow[i] * sf[i] + rs;
      #pragma unroll
      for (int nt = 0; nt < 4; ++nt) o[nt][i] *= sf[i];
    }
    // re-layout P (C layout) -> A layout via wave-private LDS
    #pragma unroll
    for (int nt = 0; nt < 4; ++nt)
      #pragma unroll
      for (int i = 0; i < 8; ++i)
        pbuf[wave][i + 8 * hi][nt * 16 + lo] = (_Float16)s[nt][i];
    const _Float16* pr = &pbuf[wave][lo][0];
    v16h ap0 = cat8(*(const v8h*)(pr + hi * 16),      *(const v8h*)(pr + 8 + hi * 16));
    v16h ap1 = cat8(*(const v8h*)(pr + 32 + hi * 16), *(const v8h*)(pr + 40 + hi * 16));
    // O += P @ V
    #pragma unroll
    for (int nt = 0; nt < 4; ++nt) {
      const _Float16* vp = vbase + (size_t)(nt * 16 + lo) * LSEQ + kb;
      v16h bv0 = *(const v16h*)(vp + hi * 16);
      v16h bv1 = *(const v16h*)(vp + 32 + hi * 16);
      o[nt] = WMMA16(ap0, bv0, o[nt]);
      o[nt] = WMMA16(ap1, bv1, o[nt]);
    }
  }
  // normalize and reduce over this wave's 16 queries, accumulate into gsum[b, h*64+d]
  float inv[8];
  #pragma unroll
  for (int i = 0; i < 8; ++i) inv[i] = 1.f / lrow[i];
  #pragma unroll
  for (int nt = 0; nt < 4; ++nt) {
    float cs = 0.f;
    #pragma unroll
    for (int i = 0; i < 8; ++i) cs += o[nt][i] * inv[i];
    cs += __shfl_xor(cs, 16);
    if (lane < 16) atomicAdd(&gsum[bh * DHEAD + nt * 16 + lo], cs);
  }
}

// ---------------- gsum -> gmod = a * (mean @ Wo^T + bo) ----------------
__global__ void outproj_kernel(const float* __restrict__ gsum, const float* __restrict__ Wo,
                               const float* __restrict__ bo, const float* __restrict__ alpha,
                               float* __restrict__ gmod) {
  int wave = threadIdx.x >> 5, lane = threadIdx.x & 31;
  int idx = blockIdx.x * 8 + wave;     // 0 .. 8*1024-1
  int bb = idx >> 10, n = idx & 1023;
  const float* wr = Wo + (size_t)n * C_DIM;
  const float* gr = gsum + bb * C_DIM;
  float acc = 0.f;
  for (int j = 0; j < 32; ++j) {
    int k = lane + 32 * j;
    acc += gr[k] * wr[k];
  }
  #pragma unroll
  for (int m = 16; m >= 1; m >>= 1) acc += __shfl_xor(acc, m);
  if (lane == 0) {
    float a = 0.3f / (1.f + __expf(-alpha[0]));
    gmod[bb * C_DIM + n] = a * (acc * (1.f / 256.f) + bo[n]);
  }
}

__global__ void zero_kernel(float* __restrict__ p, int n) {
  int i = blockIdx.x * blockDim.x + threadIdx.x;
  if (i < n) p[i] = 0.f;
}

__global__ void finaladd_kernel(const float* __restrict__ X, const float* __restrict__ gmod,
                                float* __restrict__ out) {
  size_t i = (size_t)blockIdx.x * blockDim.x + threadIdx.x;  // float4 index
  float4 v = ((const float4*)X)[i];
  size_t b = i / ((size_t)LSEQ * (C_DIM / 4));
  int c4 = (int)(i & (C_DIM / 4 - 1));
  float4 g = ((const float4*)gmod)[b * (C_DIM / 4) + c4];
  v.x += g.x; v.y += g.y; v.z += g.z; v.w += g.w;
  ((float4*)out)[i] = v;
}

extern "C" void kernel_launch(void* const* d_in, const int* in_sizes, int n_in,
                              void* d_out, int out_size, void* d_ws, size_t ws_size,
                              hipStream_t stream) {
  (void)in_sizes; (void)n_in; (void)out_size; (void)ws_size;
  const float* prompt = (const float*)d_in[0];
  const float* x      = (const float*)d_in[1];
  const float* ln_q_w = (const float*)d_in[2];
  const float* ln_q_b = (const float*)d_in[3];
  const float* ln_k_w = (const float*)d_in[4];
  const float* ln_k_b = (const float*)d_in[5];
  const float* Wq     = (const float*)d_in[6];
  const float* bq     = (const float*)d_in[7];
  const float* Wk     = (const float*)d_in[8];
  const float* bk     = (const float*)d_in[9];
  const float* Wo     = (const float*)d_in[10];
  const float* bo     = (const float*)d_in[11];
  const float* alpha  = (const float*)d_in[12];
  float* out = (float*)d_out;

  char* ws = (char*)d_ws;
  size_t off = 0;
  auto alloc = [&](size_t bytes) -> void* {
    void* p = ws + off;
    off += bytes;
    off = (off + 255) & ~(size_t)255;
    return p;
  };
  float*    statsX = (float*)alloc((size_t)BATCH * LSEQ * 2 * 4);
  float*    statsP = (float*)alloc((size_t)BATCH * LP * 2 * 4);
  _Float16* Wq16   = (_Float16*)alloc((size_t)C_DIM * C_DIM * 2);
  _Float16* Wk16   = (_Float16*)alloc((size_t)C_DIM * C_DIM * 2);
  float*    biasQ  = (float*)alloc(C_DIM * 4);
  float*    biasK  = (float*)alloc(C_DIM * 4);
  _Float16* Q16    = (_Float16*)alloc((size_t)BATCH * LP * C_DIM * 2);
  _Float16* K16    = (_Float16*)alloc((size_t)BATCH * LSEQ * C_DIM * 2);
  _Float16* VT16   = (_Float16*)alloc((size_t)BATCH * LSEQ * C_DIM * 2);
  float*    gsum   = (float*)alloc((size_t)BATCH * C_DIM * 4);
  float*    gmod   = (float*)alloc((size_t)BATCH * C_DIM * 4);

  rowstats_kernel<<<BATCH * LSEQ / 8, 256, 0, stream>>>(x, statsX, BATCH * LSEQ);
  rowstats_kernel<<<BATCH * LP / 8, 256, 0, stream>>>(prompt, statsP, BATCH * LP);
  packw_kernel<<<C_DIM / 8, 256, 0, stream>>>(Wq, ln_q_w, ln_q_b, bq, Wq16, biasQ);
  packw_kernel<<<C_DIM / 8, 256, 0, stream>>>(Wk, ln_k_w, ln_k_b, bk, Wk16, biasK);
  packv_kernel<<<BATCH * NHEAD * (LSEQ / 64), 256, 0, stream>>>(x, VT16);
  lngemm_kernel<<<(BATCH * LP / 128) * 8, 256, 0, stream>>>(prompt, statsP, Wq16, biasQ, Q16);
  lngemm_kernel<<<(BATCH * LSEQ / 128) * 8, 256, 0, stream>>>(x, statsX, Wk16, biasK, K16);
  zero_kernel<<<(BATCH * C_DIM + 255) / 256, 256, 0, stream>>>(gsum, BATCH * C_DIM);
  attn_kernel<<<BATCH * NHEAD * 2, 256, 0, stream>>>(Q16, K16, VT16, gsum);
  outproj_kernel<<<BATCH * C_DIM / 8, 256, 0, stream>>>(gsum, Wo, bo, alpha, gmod);
  finaladd_kernel<<<BATCH * LSEQ * (C_DIM / 4) / 256, 256, 0, stream>>>(x, gmod, out);
}
